// MoD_74732430950687
// MI455X (gfx1250) — matrix operand order repaired and checked by reading, not compiled, not run on hardware
//
#include <hip/hip_runtime.h>
#include <cstdint>
#include <cstddef>

#define BB   4
#define SS   4096
#define DD   2048
#define HH   16
#define HDD  128
#define HIDD 5632
#define KSEL 491
#define KP   512
#define MP   (BB*KP)   /* 2048 padded rows */

#define LDS_STRIDE 40  /* 32 K-elems + 8 pad ushorts per tile row (80B, 16B aligned) */

#define AS1 __attribute__((address_space(1)))
#define AS3 __attribute__((address_space(3)))

#if defined(__has_builtin)
#if __has_builtin(__builtin_amdgcn_global_load_async_to_lds_b128)
#define HAVE_ASYNC_LDS 1
#endif
#endif

typedef __attribute__((ext_vector_type(16))) __bf16 v16bf;
typedef __attribute__((ext_vector_type(8)))  float  v8f;
typedef __attribute__((ext_vector_type(4)))  int    v4i;

__device__ __forceinline__ unsigned short f2bf(float x) {
  unsigned u = __float_as_uint(x);
  u += 0x7FFFu + ((u >> 16) & 1u);          // round-to-nearest-even
  return (unsigned short)(u >> 16);
}

// ---------------------------------------------------------------------------
// WMMA fragment loads from LDS tiles (layouts per CDNA5 ISA)
// As: rows x 32K, row stride LDS_STRIDE. Bs: cols x 32K (pre-transposed).
// ---------------------------------------------------------------------------
__device__ __forceinline__ v16bf load_a_frag(const unsigned short* As, int m0, int lane) {
  int m  = m0 + (lane & 15);
  int hi = lane >> 4;                        // 0: K 0..7,16..23  1: K 8..15,24..31
  const unsigned* row = (const unsigned*)(As + m * LDS_STRIDE);
  union { v16bf v; unsigned u[8]; } f;
#pragma unroll
  for (int j = 0; j < 4; ++j) {
    f.u[j]     = row[hi * 4 + j];            // K = 8*hi + 2j, +1
    f.u[j + 4] = row[8 + hi * 4 + j];        // K = 16 + 8*hi + 2j, +1
  }
  return f.v;
}

__device__ __forceinline__ v16bf load_b_frag(const unsigned short* Bs, int n0, int lane) {
  int n = n0 + (lane & 15);
  const unsigned* col = (const unsigned*)(Bs + n * LDS_STRIDE);
  int hi = lane >> 4;                        // 0: K 0..7,16..23  1: K 8..15,24..31
  union { v16bf v; unsigned u[8]; } f;
#pragma unroll
  for (int j = 0; j < 4; ++j) { f.u[j] = col[hi * 4 + j]; f.u[j + 4] = col[8 + hi * 4 + j]; }
  return f.v;
}

__device__ __forceinline__ void wait_stage() {
#if HAVE_ASYNC_LDS
#if __has_builtin(__builtin_amdgcn_s_wait_asynccnt)
  __builtin_amdgcn_s_wait_asynccnt(0);
#else
  asm volatile("s_wait_asynccnt 0x0" ::: "memory");
#endif
#endif
  __syncthreads();
}

#define WMMA_BF16(a, b, c) \
  __builtin_amdgcn_wmma_f32_16x16x32_bf16(false, (a), false, (b), (short)0, (c), false, false)

// ---------------------------------------------------------------------------
// bf16 WMMA GEMM: C[M,N] = A[M,Kd] * Bt^T, Bt stored [N][Kd] (pre-transposed).
// 256 threads = 8 waves; block tile 128x128; each wave 32x64 (2x4 wmma tiles).
// Double-buffered LDS; tiles staged with GLOBAL_LOAD_ASYNC_TO_LDS_B128
// (ASYNCcnt) when available, overlapped with WMMA on the other buffer.
// Batched via z = zb*zdiv + zh with separate strides (per-(b,h) attention).
// ---------------------------------------------------------------------------
__global__ __launch_bounds__(256)
void k_gemm(const unsigned short* __restrict__ A, long lda, long sAb, long sAh,
            const unsigned short* __restrict__ Bt, long ldb, long sBb, long sBh,
            float* __restrict__ C, long ldc, long sCb, long sCh,
            int zdiv, int Kd)
{
  __shared__ unsigned short As[2][128 * LDS_STRIDE];
  __shared__ unsigned short Bs[2][128 * LDS_STRIDE];

  int z  = blockIdx.z;
  int zb = z / zdiv, zh = z % zdiv;
  A  += (long)zb * sAb + (long)zh * sAh;
  Bt += (long)zb * sBb + (long)zh * sBh;
  C  += (long)zb * sCb + (long)zh * sCh;

  int tileM = blockIdx.y * 128, tileN = blockIdx.x * 128;
  int tid = threadIdx.x, wave = tid >> 5, lane = tid & 31;
  int wm = (wave & 3) * 32, wn = (wave >> 2) * 64;

  const unsigned short* gA = A  + (long)tileM * lda;
  const unsigned short* gB = Bt + (long)tileN * ldb;

  // stage one 128x32 A tile + 128x32 B tile: 512 16-byte chunks each,
  // each thread moves chunks tid and tid+256 of both matrices.
  auto stage = [&](int buf, int k0) {
#pragma unroll
    for (int c = 0; c < 2; ++c) {
      int ch  = tid + c * 256;
      int row = ch >> 2, q = ch & 3;
      const unsigned short* ga = gA + (long)row * lda + k0 + q * 8;
      const unsigned short* gb = gB + (long)row * ldb + k0 + q * 8;
      unsigned short* la = &As[buf][row * LDS_STRIDE + q * 8];
      unsigned short* lb = &Bs[buf][row * LDS_STRIDE + q * 8];
#if HAVE_ASYNC_LDS
      __builtin_amdgcn_global_load_async_to_lds_b128((AS1 v4i*)ga, (AS3 v4i*)la, 0, 0);
      __builtin_amdgcn_global_load_async_to_lds_b128((AS1 v4i*)gb, (AS3 v4i*)lb, 0, 0);
#else
      *(uint4*)la = *(const uint4*)ga;
      *(uint4*)lb = *(const uint4*)gb;
#endif
    }
  };

  v8f acc[2][4] = {};

  stage(0, 0);
  wait_stage();
  int cur = 0;
  for (int k0 = 0; k0 < Kd; k0 += 32) {
    if (k0 + 32 < Kd) stage(cur ^ 1, k0 + 32);   // overlap next tile with WMMA

    v16bf a0 = load_a_frag(As[cur], wm + 0,  lane);
    v16bf a1 = load_a_frag(As[cur], wm + 16, lane);
    v16bf b0 = load_b_frag(Bs[cur], wn + 0,  lane);
    v16bf b1 = load_b_frag(Bs[cur], wn + 16, lane);
    v16bf b2 = load_b_frag(Bs[cur], wn + 32, lane);
    v16bf b3 = load_b_frag(Bs[cur], wn + 48, lane);

    acc[0][0] = WMMA_BF16(a0, b0, acc[0][0]);
    acc[0][1] = WMMA_BF16(a0, b1, acc[0][1]);
    acc[0][2] = WMMA_BF16(a0, b2, acc[0][2]);
    acc[0][3] = WMMA_BF16(a0, b3, acc[0][3]);
    acc[1][0] = WMMA_BF16(a1, b0, acc[1][0]);
    acc[1][1] = WMMA_BF16(a1, b1, acc[1][1]);
    acc[1][2] = WMMA_BF16(a1, b2, acc[1][2]);
    acc[1][3] = WMMA_BF16(a1, b3, acc[1][3]);

    wait_stage();
    cur ^= 1;
  }

  // C layout: lane (l&15) = N, VGPR r -> M = r + 8*(l>=16)
  int nl = lane & 15, mh = (lane >> 4) * 8;
#pragma unroll
  for (int i = 0; i < 2; ++i)
#pragma unroll
    for (int j = 0; j < 4; ++j) {
      int mb = tileM + wm + i * 16 + mh;
      int nb = tileN + wn + j * 16 + nl;
#pragma unroll
      for (int r = 0; r < 8; ++r)
        C[(long)(mb + r) * ldc + nb] = acc[i][j][r];
    }
}

// ---------------------------------------------------------------------------
// fp32 [rows,cols] -> bf16 transposed [cols,rows] (coalesced via LDS tile)
// batched with z = zb*zdiv + zh strides (for per-(b,h) V transpose)
// ---------------------------------------------------------------------------
__global__ __launch_bounds__(256)
void k_transpose_bf16(const float* __restrict__ src, long sld, long sB, long sH,
                      unsigned short* __restrict__ dst, long dld, long dB, long dH,
                      int zdiv) {
  __shared__ float tile[32][33];
  int z = blockIdx.z, zb = z / zdiv, zh = z % zdiv;
  src += (long)zb * sB + (long)zh * sH;
  dst += (long)zb * dB + (long)zh * dH;
  int r0 = blockIdx.y * 32, c0 = blockIdx.x * 32;
  int tr = threadIdx.x >> 5, tc = threadIdx.x & 31;
#pragma unroll
  for (int i = 0; i < 32; i += 8)
    tile[tr + i][tc] = src[(long)(r0 + tr + i) * sld + c0 + tc];
  __syncthreads();
#pragma unroll
  for (int i = 0; i < 32; i += 8)
    dst[(long)(c0 + tr + i) * dld + r0 + tc] = f2bf(tile[tc][tr + i]);
}

// ---------------------------------------------------------------------------
// fp32 -> bf16 convert
// ---------------------------------------------------------------------------
__global__ void k_f32_to_bf16(const float* __restrict__ src, unsigned short* __restrict__ dst, long n) {
  long i = (long)blockIdx.x * blockDim.x + threadIdx.x;
  if (i < n) dst[i] = f2bf(src[i]);
}

// ---------------------------------------------------------------------------
// router + aux logits: one wave per token, two dot products of length D
// ---------------------------------------------------------------------------
__global__ __launch_bounds__(256)
void k_router(const float* __restrict__ x, const float* __restrict__ wr,
              const float* __restrict__ wa, float* __restrict__ lr, float* __restrict__ la) {
  int token = blockIdx.x * 8 + (threadIdx.x >> 5);
  int lane  = threadIdx.x & 31;
  if (token >= BB * SS) return;
  const float* xp = x + (long)token * DD;
  float sr = 0.f, sa = 0.f;
  for (int d = lane; d < DD; d += 32) { float xv = xp[d]; sr += xv * wr[d]; sa += xv * wa[d]; }
  for (int off = 16; off; off >>= 1) { sr += __shfl_down(sr, off, 32); sa += __shfl_down(sa, off, 32); }
  if (lane == 0) { lr[token] = sr; la[token] = sa; }
}

// ---------------------------------------------------------------------------
// per-batch top-K: bitonic sort 4096 by (val desc, idx asc), then re-sort the
// top 491 by index ascending (padded to 512 with sentinel).
// ---------------------------------------------------------------------------
__global__ __launch_bounds__(1024)
void k_topk(const float* __restrict__ lr, int* __restrict__ sel, float* __restrict__ rw) {
  __shared__ float v[4096];
  __shared__ int   id[4096];
  __shared__ float v2[512];
  __shared__ int   id2[512];
  int b = blockIdx.x, tid = threadIdx.x;
  for (int i = tid; i < 4096; i += 1024) { v[i] = lr[b * 4096 + i]; id[i] = i; }
  __syncthreads();
  for (int ksz = 2; ksz <= 4096; ksz <<= 1)
    for (int j = ksz >> 1; j > 0; j >>= 1) {
      for (int t = tid; t < 4096; t += 1024) {
        int ixj = t ^ j;
        if (ixj > t) {
          float va = v[t], vb = v[ixj]; int ia = id[t], ib = id[ixj];
          bool aBefore = (va > vb) || (va == vb && ia < ib);
          bool up = ((t & ksz) == 0);
          if (up ? !aBefore : aBefore) { v[t] = vb; v[ixj] = va; id[t] = ib; id[ixj] = ia; }
        }
      }
      __syncthreads();
    }
  for (int i = tid; i < 512; i += 1024) {
    if (i < KSEL) { v2[i] = v[i]; id2[i] = id[i]; }
    else          { v2[i] = 0.f;  id2[i] = 0x7FFFFFFF; }
  }
  __syncthreads();
  for (int ksz = 2; ksz <= 512; ksz <<= 1)
    for (int j = ksz >> 1; j > 0; j >>= 1) {
      for (int t = tid; t < 512; t += 1024) {
        int ixj = t ^ j;
        if (ixj > t) {
          bool aBefore = id2[t] < id2[ixj];
          bool up = ((t & ksz) == 0);
          if (up ? !aBefore : aBefore) {
            float tv = v2[t]; v2[t] = v2[ixj]; v2[ixj] = tv;
            int   ti = id2[t]; id2[t] = id2[ixj]; id2[ixj] = ti;
          }
        }
      }
      __syncthreads();
    }
  for (int i = tid; i < 512; i += 1024) {
    sel[b * 512 + i] = (i < KSEL) ? id2[i] : -1;
    rw[b * 512 + i]  = (i < KSEL) ? v2[i]  : 0.f;
  }
}

// ---------------------------------------------------------------------------
// gather selected tokens + RMSNorm -> fx (fp32 residual) and h (bf16)
// ---------------------------------------------------------------------------
__global__ __launch_bounds__(256)
void k_gather_norm(const float* __restrict__ x, const int* __restrict__ sel,
                   const float* __restrict__ g1, float* __restrict__ fx,
                   unsigned short* __restrict__ hb) {
  int r = blockIdx.x, b = r >> 9, kk = r & 511;
  int s = sel[b * 512 + kk];
  float* fxr = fx + (long)r * DD;
  unsigned short* hr = hb + (long)r * DD;
  if (s < 0) {
    for (int d = threadIdx.x; d < DD; d += 256) { fxr[d] = 0.f; hr[d] = 0; }
    return;
  }
  __shared__ float red[8];
  __shared__ float bro;
  const float* xp = x + ((long)b * SS + s) * DD;
  float ss = 0.f;
  for (int d = threadIdx.x; d < DD; d += 256) { float xv = xp[d]; ss += xv * xv; }
  for (int off = 16; off; off >>= 1) ss += __shfl_down(ss, off, 32);
  if ((threadIdx.x & 31) == 0) red[threadIdx.x >> 5] = ss;
  __syncthreads();
  if (threadIdx.x == 0) { float t = 0.f; for (int i = 0; i < 8; ++i) t += red[i]; bro = rsqrtf(t / (float)DD + 1e-5f); }
  __syncthreads();
  float rms = bro;
  for (int d = threadIdx.x; d < DD; d += 256) { float xv = xp[d]; fxr[d] = xv; hr[d] = f2bf(xv * g1[d] * rms); }
}

// ---------------------------------------------------------------------------
// residual add + RMSNorm: fx2 = a + o; h2 = bf16(rmsnorm(fx2, g))
// ---------------------------------------------------------------------------
__global__ __launch_bounds__(256)
void k_resid_norm(const float* __restrict__ a, const float* __restrict__ o,
                  const float* __restrict__ g, float* __restrict__ fx2,
                  unsigned short* __restrict__ hb) {
  int r = blockIdx.x;
  const float* ar = a + (long)r * DD;
  const float* orr = o + (long)r * DD;
  float* f2 = fx2 + (long)r * DD;
  unsigned short* hr = hb + (long)r * DD;
  __shared__ float red[8];
  __shared__ float bro;
  float ss = 0.f;
  for (int d = threadIdx.x; d < DD; d += 256) { float v = ar[d] + orr[d]; f2[d] = v; ss += v * v; }
  for (int off = 16; off; off >>= 1) ss += __shfl_down(ss, off, 32);
  if ((threadIdx.x & 31) == 0) red[threadIdx.x >> 5] = ss;
  __syncthreads();
  if (threadIdx.x == 0) { float t = 0.f; for (int i = 0; i < 8; ++i) t += red[i]; bro = rsqrtf(t / (float)DD + 1e-5f); }
  __syncthreads();
  float rms = bro;
  for (int d = threadIdx.x; d < DD; d += 256) hr[d] = f2bf(f2[d] * g[d] * rms);
}

// ---------------------------------------------------------------------------
// RoPE on q and k (fp32 in, bf16 out); pad rows are zero and stay zero
// ---------------------------------------------------------------------------
__global__ void k_rope(const float* __restrict__ qf, unsigned short* __restrict__ qb,
                       const float* __restrict__ kf, unsigned short* __restrict__ kb) {
  long i = (long)blockIdx.x * blockDim.x + threadIdx.x;
  if (i >= (long)MP * HH * 64) return;
  int dp = (int)(i & 63);
  long rh = i >> 6;
  int h = (int)(rh & 15);
  int r = (int)(rh >> 4);
  float pos = (float)(r & 511);
  float freq = __powf(10000.f, -(float)dp / 64.f);
  float ang = pos * freq;
  float c = __cosf(ang), sn = __sinf(ang);
  long base = ((long)r * HH + h) * HDD + dp;
  float x1 = qf[base], x2 = qf[base + 64];
  qb[base]      = f2bf(x1 * c - x2 * sn);
  qb[base + 64] = f2bf(x1 * sn + x2 * c);
  x1 = kf[base]; x2 = kf[base + 64];
  kb[base]      = f2bf(x1 * c - x2 * sn);
  kb[base + 64] = f2bf(x1 * sn + x2 * c);
}

// ---------------------------------------------------------------------------
// causal softmax over one score row (scale applied here); bf16 probs out
// ---------------------------------------------------------------------------
__global__ __launch_bounds__(256)
void k_softmax(const float* __restrict__ scores, unsigned short* __restrict__ attn) {
  int row = blockIdx.x;
  int i = row & 511;
  const float* sr = scores + (long)row * KP;
  unsigned short* ar = attn + (long)row * KP;
  __shared__ float red[8];
  __shared__ float bro;
  const float scale = 0.08838834764831845f;   // 1/sqrt(128)
  int nvalid = (i < KSEL) ? (i + 1) : 0;
  float mx = -1e30f;
  for (int j = threadIdx.x; j < KP; j += 256)
    if (j < nvalid) mx = fmaxf(mx, sr[j] * scale);
  for (int off = 16; off; off >>= 1) mx = fmaxf(mx, __shfl_down(mx, off, 32));
  if ((threadIdx.x & 31) == 0) red[threadIdx.x >> 5] = mx;
  __syncthreads();
  if (threadIdx.x == 0) { float m = -1e30f; for (int t = 0; t < 8; ++t) m = fmaxf(m, red[t]); bro = m; }
  __syncthreads();
  mx = bro;
  float e[2]; float sum = 0.f;
  for (int j = threadIdx.x, q = 0; j < KP; j += 256, ++q) {
    float v = (j < nvalid) ? __expf(sr[j] * scale - mx) : 0.f;
    e[q] = v; sum += v;
  }
  for (int off = 16; off; off >>= 1) sum += __shfl_down(sum, off, 32);
  if ((threadIdx.x & 31) == 0) red[threadIdx.x >> 5] = sum;
  __syncthreads();
  if (threadIdx.x == 0) { float t = 0.f; for (int q = 0; q < 8; ++q) t += red[q]; bro = (t > 0.f) ? 1.f / t : 0.f; }
  __syncthreads();
  float inv = bro;
  for (int j = threadIdx.x, q = 0; j < KP; j += 256, ++q) ar[j] = f2bf(e[q] * inv);
}

// ---------------------------------------------------------------------------
// silu(gate) * up -> bf16
// ---------------------------------------------------------------------------
__global__ void k_silu(const float* __restrict__ g, const float* __restrict__ u,
                       unsigned short* __restrict__ s, long n) {
  long i = (long)blockIdx.x * blockDim.x + threadIdx.x;
  if (i < n) {
    float gv = g[i];
    float sv = gv / (1.f + __expf(-gv));
    s[i] = f2bf(sv * u[i]);
  }
}

// ---------------------------------------------------------------------------
// scatter-add: out[b, sel, :] += rw * (fx2 + ffo)   (sel unique per batch)
// ---------------------------------------------------------------------------
__global__ __launch_bounds__(256)
void k_scatter(const float* __restrict__ fx2, const float* __restrict__ ffo,
               const int* __restrict__ sel, const float* __restrict__ rw,
               float* __restrict__ out) {
  int r = blockIdx.x, b = r >> 9, kk = r & 511;
  int s = sel[b * 512 + kk];
  if (s < 0) return;
  float w = rw[b * 512 + kk];
  const float* a = fx2 + (long)r * DD;
  const float* f = ffo + (long)r * DD;
  float* op = out + ((long)b * SS + s) * DD;
  for (int d = threadIdx.x; d < DD; d += 256) op[d] += w * (a[d] + f[d]);
}

// ---------------------------------------------------------------------------
// aux loss helpers
// ---------------------------------------------------------------------------
__global__ void k_flags_zero(int* flags) {
  int i = blockIdx.x * blockDim.x + threadIdx.x;
  if (i < BB * SS) flags[i] = 0;
}
__global__ void k_flags_set(const int* __restrict__ sel, int* __restrict__ flags) {
  int i = blockIdx.x * blockDim.x + threadIdx.x;
  if (i < BB * 512) { int s = sel[i]; if (s >= 0) flags[(i >> 9) * SS + s] = 1; }
}
__global__ void k_aux_zero(float* aux) { if (threadIdx.x == 0 && blockIdx.x == 0) *aux = 0.f; }
__global__ __launch_bounds__(256)
void k_aux(const float* __restrict__ la, const int* __restrict__ flags, float* __restrict__ aux) {
  __shared__ float red[8];
  int i = blockIdx.x * 256 + threadIdx.x;
  float term = 0.f;
  if (i < BB * SS) {
    float pv = 1.f / (1.f + __expf(-la[i]));
    pv = fminf(fmaxf(pv, 1e-7f), 1.f - 1e-7f);
    float tgt = flags[i] ? 1.f : 0.f;
    term = -(tgt * __logf(pv) + (1.f - tgt) * __logf(1.f - pv)) / (float)(BB * SS);
  }
  for (int off = 16; off; off >>= 1) term += __shfl_down(term, off, 32);
  if ((threadIdx.x & 31) == 0) red[threadIdx.x >> 5] = term;
  __syncthreads();
  if (threadIdx.x == 0) { float t = 0.f; for (int q = 0; q < 8; ++q) t += red[q]; atomicAdd(aux, t); }
}

// ===========================================================================
extern "C" void kernel_launch(void* const* d_in, const int* in_sizes, int n_in,
                              void* d_out, int out_size, void* d_ws, size_t ws_size,
                              hipStream_t stream) {
  const float* x  = (const float*)d_in[0];
  const float* wr = (const float*)d_in[1];
  const float* wa = (const float*)d_in[2];
  const float* wq = (const float*)d_in[3];
  const float* wk = (const float*)d_in[4];
  const float* wv = (const float*)d_in[5];
  const float* wo = (const float*)d_in[6];
  const float* wg = (const float*)d_in[7];
  const float* wu = (const float*)d_in[8];
  const float* wd = (const float*)d_in[9];
  const float* g1 = (const float*)d_in[10];
  const float* g2 = (const float*)d_in[11];

  float* out = (float*)d_out;
  float* aux = out + (size_t)BB * SS * DD;

  // ---- workspace bump allocator -------------------------------------------
  char* p = (char*)d_ws;
  auto alloc = [&](size_t bytes) -> void* {
    void* r = (void*)p;
    p += (bytes + 255) & ~(size_t)255;
    return r;
  };
  float* logit_r = (float*)alloc((size_t)BB * SS * 4);
  float* logit_a = (float*)alloc((size_t)BB * SS * 4);
  int*   sel     = (int*)  alloc((size_t)BB * 512 * 4);
  float* rw      = (float*)alloc((size_t)BB * 512 * 4);
  int*   flags   = (int*)  alloc((size_t)BB * SS * 4);
  unsigned short* wqT = (unsigned short*)alloc((size_t)DD * DD * 2);     // [N][K]
  unsigned short* wkT = (unsigned short*)alloc((size_t)DD * DD * 2);
  unsigned short* wvT = (unsigned short*)alloc((size_t)DD * DD * 2);
  unsigned short* woT = (unsigned short*)alloc((size_t)DD * DD * 2);
  unsigned short* wgT = (unsigned short*)alloc((size_t)HIDD * DD * 2);   // [HID][D]
  unsigned short* wuT = (unsigned short*)alloc((size_t)HIDD * DD * 2);
  unsigned short* wdT = (unsigned short*)alloc((size_t)DD * HIDD * 2);   // [D][HID]
  float* fx    = (float*)alloc((size_t)MP * DD * 4);
  unsigned short* h_bf = (unsigned short*)alloc((size_t)MP * DD * 2);
  float* q_f = (float*)alloc((size_t)MP * DD * 4);   // q_f,k_f,v_f contiguous:
  float* k_f = (float*)alloc((size_t)MP * DD * 4);   //   48 MB, reused as u_f (44 MB)
  float* v_f = (float*)alloc((size_t)MP * DD * 4);
  unsigned short* q_bf = (unsigned short*)alloc((size_t)MP * DD * 2);
  unsigned short* k_bf = (unsigned short*)alloc((size_t)MP * DD * 2);
  unsigned short* vT   = (unsigned short*)alloc((size_t)BB * HH * HDD * KP * 2); // [b][h][d][j]
  float* scores = (float*)alloc((size_t)BB * HH * KP * KP * 4);   // reused as g_f
  unsigned short* attn = (unsigned short*)alloc((size_t)BB * HH * KP * KP * 2); // reused as s_bf
  float* o_f   = (float*)alloc((size_t)MP * DD * 4);
  unsigned short* o_bf = (unsigned short*)alloc((size_t)MP * DD * 2);
  float* oproj = (float*)alloc((size_t)MP * DD * 4);
  float* fx2   = (float*)alloc((size_t)MP * DD * 4);
  unsigned short* h2_bf = (unsigned short*)alloc((size_t)MP * DD * 2);
  float* ffo   = (float*)alloc((size_t)MP * DD * 4);
  float* g_f = scores;                  // reuse (scores dead after softmax)
  float* u_f = q_f;                     // reuse (q/k/v fp32 dead after rope)
  unsigned short* s_bf = attn;          // reuse (attn dead after attn.V)

  auto gemm = [&](const unsigned short* A, long lda, long sAb, long sAh,
                  const unsigned short* Bt, long ldb, long sBb, long sBh,
                  float* C, long ldc, long sCb, long sCh,
                  int zdiv, int Z, int M, int N, int Kd) {
    dim3 grid((unsigned)(N / 128), (unsigned)(M / 128), (unsigned)Z);
    k_gemm<<<grid, dim3(256), 0, stream>>>(A, lda, sAb, sAh, Bt, ldb, sBb, sBh,
                                           C, ldc, sCb, sCh, zdiv, Kd);
  };
  auto transp = [&](const float* src, long sld, long sB, long sH,
                    unsigned short* dst, long dld, long dB, long dH,
                    int rows, int cols, int zdiv, int Z) {
    dim3 grid((unsigned)(cols / 32), (unsigned)(rows / 32), (unsigned)Z);
    k_transpose_bf16<<<grid, dim3(256), 0, stream>>>(src, sld, sB, sH, dst, dld, dB, dH, zdiv);
  };

  // 1) out = x ; aux = 0
  (void)hipMemcpyAsync(out, x, (size_t)BB * SS * DD * sizeof(float),
                       hipMemcpyDeviceToDevice, stream);
  k_aux_zero<<<dim3(1), dim3(32), 0, stream>>>(aux);

  // 2) router + aux logits, top-K selection
  k_router<<<dim3(BB * SS / 8), dim3(256), 0, stream>>>(x, wr, wa, logit_r, logit_a);
  k_topk<<<dim3(BB), dim3(1024), 0, stream>>>(logit_r, sel, rw);

  // 3) aux BCE loss
  k_flags_zero<<<dim3(BB * SS / 256), dim3(256), 0, stream>>>(flags);
  k_flags_set<<<dim3(BB * 512 / 256), dim3(256), 0, stream>>>(sel, flags);
  k_aux<<<dim3(BB * SS / 256), dim3(256), 0, stream>>>(logit_a, flags, aux);

  // 4) weights -> transposed bf16 [N][K]
  transp(wq, DD,   0, 0, wqT, DD,   0, 0, DD,   DD,   1, 1);
  transp(wk, DD,   0, 0, wkT, DD,   0, 0, DD,   DD,   1, 1);
  transp(wv, DD,   0, 0, wvT, DD,   0, 0, DD,   DD,   1, 1);
  transp(wo, DD,   0, 0, woT, DD,   0, 0, DD,   DD,   1, 1);
  transp(wg, HIDD, 0, 0, wgT, DD,   0, 0, DD,   HIDD, 1, 1);
  transp(wu, HIDD, 0, 0, wuT, DD,   0, 0, DD,   HIDD, 1, 1);
  transp(wd, DD,   0, 0, wdT, HIDD, 0, 0, HIDD, DD,   1, 1);

  // 5) gather + rmsnorm
  k_gather_norm<<<dim3(MP), dim3(256), 0, stream>>>(x, sel, g1, fx, h_bf);

  // 6) Q/K/V projections (WMMA)
  gemm(h_bf, DD, 0, 0, wqT, DD, 0, 0, q_f, DD, 0, 0, 1, 1, MP, DD, DD);
  gemm(h_bf, DD, 0, 0, wkT, DD, 0, 0, k_f, DD, 0, 0, 1, 1, MP, DD, DD);
  gemm(h_bf, DD, 0, 0, wvT, DD, 0, 0, v_f, DD, 0, 0, 1, 1, MP, DD, DD);

  // 7) RoPE(q,k) -> bf16 ; v -> per-(b,h) transposed bf16 [HD][KP]
  k_rope<<<dim3((unsigned)((long)MP * HH * 64 / 256)), dim3(256), 0, stream>>>(q_f, q_bf, k_f, k_bf);
  transp(v_f, DD, (long)KP * DD, HDD,
         vT, KP, (long)HH * HDD * KP, (long)HDD * KP,
         KP, HDD, HH, BB * HH);

  // 8) scores = q . k^T per (b,h)  (WMMA; K rows already give Bt=[N][K])
  gemm(q_bf, DD, (long)KP * DD, HDD,
       k_bf, DD, (long)KP * DD, HDD,
       scores, KP, (long)HH * KP * KP, (long)KP * KP,
       HH, BB * HH, KP, KP, HDD);

  // 9) causal softmax -> bf16 probs
  k_softmax<<<dim3(BB * HH * KP), dim3(256), 0, stream>>>(scores, attn);

  // 10) o = attn . v per (b,h)  (WMMA, vT is Bt=[N][K])
  gemm(attn, KP, (long)HH * KP * KP, (long)KP * KP,
       vT, KP, (long)HH * HDD * KP, (long)HDD * KP,
       o_f, DD, (long)KP * DD, HDD,
       HH, BB * HH, KP, HDD, KP);

  // 11) output projection + residual + norm
  k_f32_to_bf16<<<dim3((unsigned)(((long)MP * DD + 255) / 256)), dim3(256), 0, stream>>>(o_f, o_bf, (long)MP * DD);
  gemm(o_bf, DD, 0, 0, woT, DD, 0, 0, oproj, DD, 0, 0, 1, 1, MP, DD, DD);
  k_resid_norm<<<dim3(MP), dim3(256), 0, stream>>>(fx, oproj, g2, fx2, h2_bf);

  // 12) FFN (WMMA)
  gemm(h2_bf, DD, 0, 0, wgT, DD, 0, 0, g_f, HIDD, 0, 0, 1, 1, MP, HIDD, DD);
  gemm(h2_bf, DD, 0, 0, wuT, DD, 0, 0, u_f, HIDD, 0, 0, 1, 1, MP, HIDD, DD);
  k_silu<<<dim3((unsigned)(((long)MP * HIDD + 255) / 256)), dim3(256), 0, stream>>>(g_f, u_f, s_bf, (long)MP * HIDD);
  gemm(s_bf, HIDD, 0, 0, wdT, HIDD, 0, 0, ffo, DD, 0, 0, 1, 1, MP, DD, HIDD);

  // 13) scatter-add back
  k_scatter<<<dim3(MP), dim3(256), 0, stream>>>(fx2, ffo, sel, rw, out);
}
